// VLLTR_52166672777670
// MI455X (gfx1250) — matrix operand order repaired and checked by reading, not compiled, not run on hardware
//
#include <hip/hip_runtime.h>
#include <hip/hip_bf16.h>
#include <math.h>

#define DIM   512
#define BQ    256
#define BK    512
#define NK    256
#define HID   2048
#define NCLS  1000
#define NPAD  1008
#define LN_EPSF   1e-5f
#define NORM_EPSF 1e-12f

typedef __attribute__((ext_vector_type(16))) __bf16 v16bf;
typedef __attribute__((ext_vector_type(8)))  __bf16 v8bf;
typedef __attribute__((ext_vector_type(8)))  float  v8f;

// ---------------------------------------------------------------------------
// WMMA NT K-loop (wave32, one 16x16 tile per wave): acc += A[16xK] * Bt[16xK]^T
// Per-lane: m = lane&15, g = lane>>4.
//  A layout: lane holds K = [g*8 .. g*8+7] and [16+g*8 .. 16+g*8+7] of row m.
//  B layout: lane holds column m, K = [g*8 .. g*8+7] and [16+g*8 .. 16+g*8+7] of Bt row m.
// Both operands load as two 16-byte vectors -> global/ds _load_b128, no packing.
// ---------------------------------------------------------------------------
static __device__ inline void wmma_kloop_nt(const __bf16* A, int lda,
                                            const __bf16* Bt, int ldb,
                                            int K, v8f& acc)
{
  const int lane = threadIdx.x & 31;
  const int g = lane >> 4;
  const int m = lane & 15;
  const __bf16* ap = A  + (size_t)m * lda + g * 8;
  const __bf16* bp = Bt + (size_t)m * ldb + g * 8;
  for (int k0 = 0; k0 < K; k0 += 32) {
    v8bf a_lo = *(const v8bf*)(ap + k0);
    v8bf a_hi = *(const v8bf*)(ap + k0 + 16);
    v8bf b_lo = *(const v8bf*)(bp + k0);
    v8bf b_hi = *(const v8bf*)(bp + k0 + 16);
    v16bf a = __builtin_shufflevector(a_lo, a_hi,
                                      0,1,2,3,4,5,6,7,8,9,10,11,12,13,14,15);
    v16bf b = __builtin_shufflevector(b_lo, b_hi,
                                      0,1,2,3,4,5,6,7,8,9,10,11,12,13,14,15);
    acc = __builtin_amdgcn_wmma_f32_16x16x32_bf16(false, a, false, b,
                                                  (short)0, acc, false, false);
  }
}

static __device__ inline void store_tile_f32(float* C, int ldc, const v8f& acc)
{
  const int lane = threadIdx.x & 31, g = lane >> 4, n = lane & 15;
#pragma unroll
  for (int i = 0; i < 8; ++i) C[(g * 8 + i) * ldc + n] = acc[i];
}

// ---------------------------------------------------------------------------
// Weight conversion with transpose: src [K][Ns] f32 -> dst [Nd][K] bf16
// (rows n >= Ns are zero-filled padding). One-time cost, enables contiguous
// per-lane B loads in every WMMA GEMM.
// ---------------------------------------------------------------------------
__global__ void k_cvt_t(const float* __restrict__ src, __bf16* __restrict__ dst,
                        int K, int Ns, int Nd)
{
  long long idx = (long long)blockIdx.x * blockDim.x + threadIdx.x;
  if (idx >= (long long)Nd * K) return;
  int n = (int)(idx / K);
  int k = (int)(idx - (long long)n * K);
  float v = (n < Ns) ? src[(size_t)k * Ns + n] : 0.f;
  dst[idx] = (__bf16)v;
}

// ---------------------------------------------------------------------------
// Query-side prep: LN(qx)->bf16, raw qx->bf16 (for MLP), qn = l2norm(qx)
// grid = BQ blocks of 256 threads
// ---------------------------------------------------------------------------
__global__ void k_q_prep(const float* __restrict__ qx,
                         const float* __restrict__ gam, const float* __restrict__ bet,
                         __bf16* __restrict__ qln, __bf16* __restrict__ qxbf,
                         float* __restrict__ qn)
{
  __shared__ float red[256];
  const int q = blockIdx.x, t = threadIdx.x;
  const float x0 = qx[q * DIM + t];
  const float x1 = qx[q * DIM + 256 + t];

  red[t] = x0 + x1; __syncthreads();
  for (int o = 128; o > 0; o >>= 1) { if (t < o) red[t] += red[t + o]; __syncthreads(); }
  const float mean = red[0] * (1.f / DIM); __syncthreads();

  const float d0 = x0 - mean, d1 = x1 - mean;
  red[t] = d0 * d0 + d1 * d1; __syncthreads();
  for (int o = 128; o > 0; o >>= 1) { if (t < o) red[t] += red[t + o]; __syncthreads(); }
  const float var = red[0] * (1.f / DIM); __syncthreads();

  const float rstd = rsqrtf(var + LN_EPSF);
  qln[q * DIM + t]       = (__bf16)(d0 * rstd * gam[t] + bet[t]);
  qln[q * DIM + 256 + t] = (__bf16)(d1 * rstd * gam[256 + t] + bet[256 + t]);
  qxbf[q * DIM + t]       = (__bf16)x0;
  qxbf[q * DIM + 256 + t] = (__bf16)x1;

  red[t] = x0 * x0 + x1 * x1; __syncthreads();
  for (int o = 128; o > 0; o >>= 1) { if (t < o) red[t] += red[t + o]; __syncthreads(); }
  const float inv = 1.f / fmaxf(sqrtf(red[0]), NORM_EPSF);
  qn[q * DIM + t]       = x0 * inv;
  qn[q * DIM + 256 + t] = x1 * inv;
}

// ---------------------------------------------------------------------------
// q = LN(qx) @ Wq (via WqT), scaled by hd^-0.5, stored bf16. 512 tiles / 64 blocks.
// ---------------------------------------------------------------------------
__global__ void k_q_proj(const __bf16* __restrict__ qln, const __bf16* __restrict__ WqT,
                         __bf16* __restrict__ qsc, float scale)
{
  const int wave = (blockIdx.x * blockDim.x + threadIdx.x) >> 5;
  const int mt = wave >> 5;          // 16 m-tiles
  const int nt = wave & 31;          // 32 n-tiles
  v8f acc = {};
  wmma_kloop_nt(qln + (size_t)mt * 16 * DIM, DIM,
                WqT + (size_t)nt * 16 * DIM, DIM, DIM, acc);
  const int lane = threadIdx.x & 31, g = lane >> 4, n = lane & 15;
#pragma unroll
  for (int i = 0; i < 8; ++i)
    qsc[(size_t)(mt * 16 + g * 8 + i) * DIM + nt * 16 + n] = (__bf16)(acc[i] * scale);
}

// ---------------------------------------------------------------------------
// Fused k-side producer. Each block owns 32 consecutive rows (all in one
// k-batch since 32 | 256):
//   1) LayerNorm rows of kx into LDS (bf16)            -> WMMA A operand
//   2) 8 waves compute the 32x512 projection (k = LN(kx) @ Wk via WkT) -> kbf
//   3) restage RAW kx rows (still in registers) through the same LDS buffer
//      and emit the bf16 transposed raw cache kxT[k][c][n] with coalesced
//      64-byte stores -> attention's value-GEMM B operand becomes contiguous.
// grid = 131072/32 = 4096 blocks of 256.
// ---------------------------------------------------------------------------
__global__ void k_k_proj(const float* __restrict__ kx,
                         const float* __restrict__ gam, const float* __restrict__ bet,
                         const __bf16* __restrict__ WkT,
                         __bf16* __restrict__ kbf, __bf16* __restrict__ kxT)
{
  __shared__ __attribute__((aligned(16))) __bf16 Aln[32 * DIM];   // 32 KB
  const int t = threadIdx.x;
  const int row = t >> 3, sub = t & 7;             // 8 threads per row, 64 elems each
  const size_t rbase = ((size_t)blockIdx.x * 32 + row) * DIM;
  const float4* src = (const float4*)(kx + rbase + sub * 64);

  float4 v[16];
  float sum = 0.f, sq = 0.f;
#pragma unroll
  for (int i = 0; i < 16; ++i) {
    float4 x = src[i];
    v[i] = x;
    sum += x.x + x.y + x.z + x.w;
    sq  += x.x * x.x + x.y * x.y + x.z * x.z + x.w * x.w;
  }
#pragma unroll
  for (int o = 1; o < 8; o <<= 1) {
    sum += __shfl_xor(sum, o, 32);
    sq  += __shfl_xor(sq,  o, 32);
  }
  const float mean = sum * (1.f / DIM);
  const float var  = sq * (1.f / DIM) - mean * mean;
  const float rstd = rsqrtf(var + LN_EPSF);

  __bf16* arow = Aln + row * DIM + sub * 64;
  const float* gg = gam + sub * 64;
  const float* bb = bet + sub * 64;
#pragma unroll
  for (int i = 0; i < 16; ++i) {
    arow[i * 4 + 0] = (__bf16)((v[i].x - mean) * rstd * gg[i * 4 + 0] + bb[i * 4 + 0]);
    arow[i * 4 + 1] = (__bf16)((v[i].y - mean) * rstd * gg[i * 4 + 1] + bb[i * 4 + 1]);
    arow[i * 4 + 2] = (__bf16)((v[i].z - mean) * rstd * gg[i * 4 + 2] + bb[i * 4 + 2]);
    arow[i * 4 + 3] = (__bf16)((v[i].w - mean) * rstd * gg[i * 4 + 3] + bb[i * 4 + 3]);
  }
  __syncthreads();

  // ---- projection GEMM: 2 m-tiles x 32 n-tiles, 8 per wave ----
  const int wave = t >> 5;
  const int lane = t & 31, g = lane >> 4, n = lane & 15;
  for (int tile = wave; tile < 64; tile += 8) {
    const int mt = tile >> 5, nt = tile & 31;
    v8f acc = {};
    wmma_kloop_nt(Aln + mt * 16 * DIM, DIM,
                  WkT + (size_t)nt * 16 * DIM, DIM, DIM, acc);
#pragma unroll
    for (int i = 0; i < 8; ++i)
      kbf[((size_t)blockIdx.x * 32 + mt * 16 + g * 8 + i) * DIM + nt * 16 + n] =
          (__bf16)acc[i];
  }
  __syncthreads();

  // ---- restage RAW kx (bf16) in the now-free LDS buffer ----
#pragma unroll
  for (int i = 0; i < 16; ++i) {
    arow[i * 4 + 0] = (__bf16)v[i].x;
    arow[i * 4 + 1] = (__bf16)v[i].y;
    arow[i * 4 + 2] = (__bf16)v[i].z;
    arow[i * 4 + 3] = (__bf16)v[i].w;
  }
  __syncthreads();

  // ---- transposed coalesced store: kxT[k][c][n0..n0+31] ----
  const int kIdx = blockIdx.x >> 3;          // 8 blocks per k-batch
  const int nb   = (blockIdx.x & 7) * 32;    // this block's n range within k
  for (int c = t; c < DIM; c += 256) {
    __bf16* dst = kxT + ((size_t)kIdx * DIM + c) * NK + nb;
    v8bf o0, o1, o2, o3;
#pragma unroll
    for (int j = 0; j < 8; ++j) {
      o0[j] = Aln[(j +  0) * DIM + c];
      o1[j] = Aln[(j +  8) * DIM + c];
      o2[j] = Aln[(j + 16) * DIM + c];
      o3[j] = Aln[(j + 24) * DIM + c];
    }
    *(v8bf*)(dst +  0) = o0;
    *(v8bf*)(dst +  8) = o1;
    *(v8bf*)(dst + 16) = o2;
    *(v8bf*)(dst + 24) = o3;
  }
}

// ---------------------------------------------------------------------------
// Fused attention: per (k, q-group of 32) block.
// scores = q @ k^T -> softmax over NK -> v-tile = P @ kx_raw (bf16 NT GEMM,
// never hits HBM) -> x2[q,k] = (v . qn) / max(||v||, eps) * exp(logit_scale)
// grid = (BK, BQ/32) = (512, 8)
// ---------------------------------------------------------------------------
__global__ void k_attn_fused(const __bf16* __restrict__ qsc,
                             const __bf16* __restrict__ kbf,
                             const __bf16* __restrict__ kxT,
                             const unsigned char* __restrict__ mask,
                             const float* __restrict__ qn,
                             const float* __restrict__ logit_scale,
                             float* __restrict__ out)
{
  __shared__ __attribute__((aligned(16))) float  S[32 * NK];  // 32 KB logits
  __shared__ __attribute__((aligned(16))) __bf16 P[32 * NK];  // 16 KB probs
  __shared__ float dotAcc[32], ssAcc[32];

  const int k = blockIdx.x;
  const int qbase = blockIdx.y * 32;
  const int t = threadIdx.x;
  const int wave = t >> 5;
  const int lane = t & 31, g = lane >> 4, nn = lane & 15;

  const __bf16* kmat = kbf + (size_t)k * NK * DIM;
  const __bf16* kxk  = kxT + (size_t)k * DIM * NK;

  // warm L2 for the transposed raw-k values needed in phase 3
  __builtin_prefetch((const void*)(kxk + (size_t)t * ((DIM * NK) / 256)), 0, 1);

  // ---- phase 1: scores = q_scaled @ k^T (2 x 16 tiles of 16x16) ----
  for (int tile = wave; tile < 32; tile += 8) {
    const int mt = tile >> 4, nt = tile & 15;
    v8f acc = {};
    wmma_kloop_nt(qsc + (size_t)(qbase + mt * 16) * DIM, DIM,
                  kmat + (size_t)nt * 16 * DIM, DIM, DIM, acc);
    store_tile_f32(S + (mt * 16) * NK + nt * 16, NK, acc);
  }
  __syncthreads();

  // ---- phase 2: masked softmax over NK, one row per thread (t < 32) ----
  if (t < 32) {
    const unsigned char* mrow = mask + k * NK;
    float* srow = S + t * NK;
    float mx = -INFINITY;
    for (int n = 0; n < NK; ++n) {
      float s = mrow[n] ? -INFINITY : srow[n];
      srow[n] = s;
      mx = fmaxf(mx, s);
    }
    float ssum = 0.f;
    for (int n = 0; n < NK; ++n) {
      float e = __expf(srow[n] - mx);
      srow[n] = e;
      ssum += e;
    }
    const float inv = 1.f / ssum;
    __bf16* prow = P + t * NK;
    for (int n = 0; n < NK; ++n) prow[n] = (__bf16)(srow[n] * inv);
    dotAcc[t] = 0.f;
    ssAcc[t]  = 0.f;
  }
  __syncthreads();

  // ---- phase 3: v-tile = P @ kx_raw via NT GEMM on kxT, fused cosine ----
  for (int tile = wave; tile < 64; tile += 8) {    // 2 m-tiles x 32 c-tiles
    const int mt = tile >> 5, ct = tile & 31;
    v8f acc = {};
    wmma_kloop_nt(P + mt * 16 * NK, NK,
                  kxk + (size_t)ct * 16 * NK, NK, NK, acc);
#pragma unroll
    for (int i = 0; i < 8; ++i) {
      const int row = mt * 16 + g * 8 + i;
      const int col = ct * 16 + nn;
      const float vv = acc[i];
      atomicAdd(&dotAcc[row], vv * qn[(size_t)(qbase + row) * DIM + col]);
      atomicAdd(&ssAcc[row], vv * vv);
    }
  }
  __syncthreads();

  if (t < 32) {
    const float x2 = dotAcc[t] / fmaxf(sqrtf(ssAcc[t]), NORM_EPSF) *
                     __expf(logit_scale[0]);
    out[(size_t)BQ * NCLS + (size_t)(qbase + t) * BK + k] = x2;
  }
}

// ---------------------------------------------------------------------------
// MLP head: h = relu(qx @ W1 + b1) bf16   (16 x 128 tiles -> 256 blocks)
// ---------------------------------------------------------------------------
__global__ void k_mlp1(const __bf16* __restrict__ qxbf, const __bf16* __restrict__ W1T,
                       const float* __restrict__ b1, __bf16* __restrict__ h)
{
  const int wave = (blockIdx.x * blockDim.x + threadIdx.x) >> 5;
  const int mt = wave >> 7;
  const int nt = wave & 127;
  v8f acc = {};
  wmma_kloop_nt(qxbf + (size_t)mt * 16 * DIM, DIM,
                W1T + (size_t)nt * 16 * DIM, DIM, DIM, acc);
  const int lane = threadIdx.x & 31, g = lane >> 4, n = lane & 15;
  const int col = nt * 16 + n;
  const float bias = b1[col];
#pragma unroll
  for (int i = 0; i < 8; ++i)
    h[(size_t)(mt * 16 + g * 8 + i) * HID + col] = (__bf16)fmaxf(acc[i] + bias, 0.f);
}

// ---------------------------------------------------------------------------
// x1 = h @ W2 + b2 (N padded 1000->1008, store-guarded). 1008 tiles -> 126 blocks.
// ---------------------------------------------------------------------------
__global__ void k_mlp2(const __bf16* __restrict__ h, const __bf16* __restrict__ W2T,
                       const float* __restrict__ b2, float* __restrict__ out)
{
  const int wave = (blockIdx.x * blockDim.x + threadIdx.x) >> 5;
  const int mt = wave / 63;
  const int nt = wave % 63;
  v8f acc = {};
  wmma_kloop_nt(h + (size_t)mt * 16 * HID, HID,
                W2T + (size_t)nt * 16 * HID, HID, HID, acc);
  const int lane = threadIdx.x & 31, g = lane >> 4, n = lane & 15;
  const int col = nt * 16 + n;
  if (col < NCLS) {                       // guard only stores; WMMA ran full-EXEC
    const float bias = b2[col];
#pragma unroll
    for (int i = 0; i < 8; ++i)
      out[(size_t)(mt * 16 + g * 8 + i) * NCLS + col] = acc[i] + bias;
  }
}

// ---------------------------------------------------------------------------
extern "C" void kernel_launch(void* const* d_in, const int* in_sizes, int n_in,
                              void* d_out, int out_size, void* d_ws, size_t ws_size,
                              hipStream_t stream)
{
  (void)in_sizes; (void)n_in; (void)out_size; (void)ws_size;

  const float* qx          = (const float*)d_in[0];
  const float* kx          = (const float*)d_in[1];
  const unsigned char* msk = (const unsigned char*)d_in[2];
  const float* logit_scale = (const float*)d_in[3];
  const float* ln_q_g      = (const float*)d_in[4];
  const float* ln_q_b      = (const float*)d_in[5];
  const float* ln_k_g      = (const float*)d_in[6];
  const float* ln_k_b      = (const float*)d_in[7];
  const float* Wq          = (const float*)d_in[8];
  const float* Wk          = (const float*)d_in[9];
  const float* W1          = (const float*)d_in[10];
  const float* b1          = (const float*)d_in[11];
  const float* W2          = (const float*)d_in[12];
  const float* b2          = (const float*)d_in[13];
  float* out = (float*)d_out;

  // workspace (~265 MB: 128 MB projected-k bf16 + 128 MB transposed raw-k bf16
  // + transposed weights + small buffers) — trivial against 432 GiB HBM.
  char* ws = (char*)d_ws;
  size_t off = 0;
  auto take = [&](size_t bytes) -> char* {
    char* p = ws + off;
    off = (off + bytes + 255) & ~(size_t)255;
    return p;
  };
  __bf16* qln  = (__bf16*)take((size_t)BQ * DIM * 2);
  __bf16* qsc  = (__bf16*)take((size_t)BQ * DIM * 2);
  __bf16* qxbf = (__bf16*)take((size_t)BQ * DIM * 2);
  float*  qn   = (float*) take((size_t)BQ * DIM * 4);
  __bf16* WqT  = (__bf16*)take((size_t)DIM * DIM * 2);
  __bf16* WkT  = (__bf16*)take((size_t)DIM * DIM * 2);
  __bf16* W1T  = (__bf16*)take((size_t)HID * DIM * 2);
  __bf16* W2T  = (__bf16*)take((size_t)NPAD * HID * 2);
  __bf16* hbf  = (__bf16*)take((size_t)BQ * HID * 2);
  __bf16* kbf  = (__bf16*)take((size_t)BK * NK * DIM * 2);
  __bf16* kxT  = (__bf16*)take((size_t)BK * DIM * NK * 2);

  // transposed bf16 weight caches
  k_cvt_t<<<(DIM * DIM + 255) / 256, 256, 0, stream>>>(Wq, WqT, DIM, DIM, DIM);
  k_cvt_t<<<(DIM * DIM + 255) / 256, 256, 0, stream>>>(Wk, WkT, DIM, DIM, DIM);
  k_cvt_t<<<(HID * DIM + 255) / 256, 256, 0, stream>>>(W1, W1T, DIM, HID, HID);
  k_cvt_t<<<(NPAD * HID + 255) / 256, 256, 0, stream>>>(W2, W2T, HID, NCLS, NPAD);

  // query prep + projection (scale = hd^-0.5 folded into q)
  k_q_prep<<<BQ, 256, 0, stream>>>(qx, ln_q_g, ln_q_b, qln, qxbf, qn);
  const float scale = 1.0f / sqrtf((float)DIM);
  k_q_proj<<<(16 * 32) / 8, 256, 0, stream>>>(qln, WqT, qsc, scale);

  // fused LN + k projection + transposed raw-k cache
  k_k_proj<<<(BK * NK) / 32, 256, 0, stream>>>(kx, ln_k_g, ln_k_b, WkT, kbf, kxT);

  // MLP head (branch 1)
  k_mlp1<<<(16 * (HID / 16)) / 8, 256, 0, stream>>>(qxbf, W1T, b1, hbf);
  k_mlp2<<<(16 * (NPAD / 16)) / 8, 256, 0, stream>>>(hbf, W2T, b2, out);

  // fused attention + cosine (branch 2)
  dim3 agrid(BK, BQ / 32);
  k_attn_fused<<<agrid, 256, 0, stream>>>(qsc, kbf, kxT, msk, qn, logit_scale, out);
}
